// LSC_Trainer_10428180595209
// MI455X (gfx1250) — matrix operand order repaired and checked by reading, not compiled, not run on hardware
//
#include <hip/hip_runtime.h>
#include <hip/hip_bf16.h>

// ---------------------------------------------------------------------------
// MI455X (gfx1250) implementation.
// - Edge-MLP deduplicated to 512 unique bond combos (160x FLOP cut vs ref).
// - Dense GEMMs on v_wmma_f32_16x16x32_f16, wave32, 16x32 tile per wave
//   (two accumulators share one A fragment -> independent WMMAs hide the
//   WMMA->WMMA hazard; halves A traffic).
// - B operands (weights) are pre-packed f32->f16 into WMMA fragment layout:
//   per lane 16 contiguous halfs per k-chunk -> two global_load_b128 instead
//   of sixteen global_load_u16 + u64 address chains.
// - Scatter-add via global_atomic_add_f32; everything else bandwidth-trivial.
// ---------------------------------------------------------------------------

typedef __attribute__((ext_vector_type(16))) _Float16 v16h;
typedef __attribute__((ext_vector_type(8)))  _Float16 v8h;
typedef __attribute__((ext_vector_type(8)))  float    v8f;

#define NN   40960   // nodes
#define EE   81920   // edges
#define GG   1024    // graphs
#define NPG  40
#define MAXN 51
#define DIN  64
#define DOUT 32
#define FDIM (MAXN * DOUT)   // 1632 = 51*32

// ---------------------------------------------------------------------------
// Pack f32 B[K,N] (row-major) into f16 WMMA B-fragment layout:
//   Bp[(((tn*(K/32))+kc)*32 + lane)*16 + i]
//   col = tn*16 + (lane&15),  k = kc*32 + (i>>3)*16 + (lane>>4)*8 + (i&7)
// One 32-byte contiguous chunk per (lane, k-chunk).
// ---------------------------------------------------------------------------
__global__ void pack_b_f16(const float* __restrict__ B, _Float16* __restrict__ Bp,
                           int K, int N)
{
    int idx = blockIdx.x * blockDim.x + threadIdx.x;
    if (idx >= K * N) return;
    int i    = idx & 15;
    int lane = (idx >> 4) & 31;
    int rest = idx >> 9;              // tn*(K/32) + kc
    int kch  = K >> 5;
    int tn   = rest / kch, kc = rest % kch;
    int col  = tn * 16 + (lane & 15);
    int k    = kc * 32 + ((i >> 3) << 4) + ((lane >> 4) << 3) + (i & 7);
    Bp[idx] = (_Float16)B[(size_t)k * N + col];
}

// ---------------------------------------------------------------------------
// WMMA GEMM: C[M,N] = act(A[M,K] @ B[K,N] + bias). A: row-major f16.
// Bp: packed fragments (above). One wave computes a 16x32 tile.
// K multiple of 32, N multiple of 32, M multiple of 16.
// ---------------------------------------------------------------------------
__global__ void gemm_wmma_f16(const _Float16* __restrict__ A,
                              const _Float16* __restrict__ Bp,
                              const float* __restrict__ bias,
                              float* __restrict__ Cf,
                              _Float16* __restrict__ Ch,
                              int M, int N, int K, int relu)
{
    const int lane = threadIdx.x & 31;
    const int wave = (blockIdx.x * blockDim.x + threadIdx.x) >> 5;
    const int nt2   = N >> 5;                 // 32-wide column blocks
    const int total = (M >> 4) * nt2;
    if (wave >= total) return;                // uniform per wave: EXEC stays full

    const int tm = wave / nt2, tn2 = wave % nt2;
    const int row0 = tm << 4, col0 = tn2 << 5;
    const int kch  = K >> 5;

    const _Float16* pa = A + (size_t)(row0 + (lane & 15)) * K + ((lane >> 4) << 3);
    const v16h* pb0 = (const v16h*)Bp + ((size_t)(2 * tn2)     * kch) * 32 + lane;
    const v16h* pb1 = (const v16h*)Bp + ((size_t)(2 * tn2 + 1) * kch) * 32 + lane;

    v8f acc0 = {}, acc1 = {};
    for (int kc = 0; kc < kch; ++kc) {
        v8h a0 = *(const v8h*)(pa);           // A: K = base+0..7
        v8h a1 = *(const v8h*)(pa + 16);      // A: K = base+16..23
        pa += 32;
        v16h a;
#pragma unroll
        for (int i = 0; i < 8; ++i) { a[i] = a0[i]; a[8 + i] = a1[i]; }

        v16h b0 = pb0[(size_t)kc * 32];       // 32B contiguous per lane
        v16h b1 = pb1[(size_t)kc * 32];

        acc0 = __builtin_amdgcn_wmma_f32_16x16x32_f16(
                   false, a, false, b0, (short)0, acc0, false, false);
        acc1 = __builtin_amdgcn_wmma_f32_16x16x32_f16(
                   false, a, false, b1, (short)0, acc1, false, false);
    }

    const int colA  = col0 + (lane & 15);
    const int colB  = colA + 16;
    const int rbase = row0 + ((lane >> 4) << 3);
    const float bv0 = bias ? bias[colA] : 0.0f;
    const float bv1 = bias ? bias[colB] : 0.0f;
#pragma unroll
    for (int i = 0; i < 8; ++i) {
        float v0 = acc0[i] + bv0;
        float v1 = acc1[i] + bv1;
        if (relu) { v0 = fmaxf(v0, 0.0f); v1 = fmaxf(v1, 0.0f); }
        const size_t r = (size_t)(rbase + i) * N;
        if (Cf) { Cf[r + colA] = v0; Cf[r + colB] = v1; }
        if (Ch) { Ch[r + colA] = (_Float16)v0; Ch[r + colB] = (_Float16)v1; }
    }
}

// ---------------------------------------------------------------------------
// Small helper kernels (bandwidth-trivial)
// ---------------------------------------------------------------------------

// atom encoder: x[n,d] = sum_c atom_emb[c, nf[n,c], d]   (N x 64, f32 + f16)
__global__ void atom_enc(const int* __restrict__ nf, const float* __restrict__ emb,
                         float* __restrict__ xf, _Float16* __restrict__ xh)
{
    int idx = blockIdx.x * blockDim.x + threadIdx.x;
    if (idx >= NN * DIN) return;
    int n = idx >> 6, d = idx & 63;
    const int* f = nf + n * 9;
    float s = 0.0f;
#pragma unroll
    for (int c = 0; c < 9; ++c) s += emb[((size_t)c * 128 + f[c]) * DIN + d];
    xf[idx] = s;
    xh[idx] = (_Float16)s;
}

// combo id per edge: 3 columns in [0,8) -> 512 combos
__global__ void edge_combo(const int* __restrict__ ef, int* __restrict__ combo)
{
    int e = blockIdx.x * blockDim.x + threadIdx.x;
    if (e < EE) combo[e] = ef[e * 3] * 64 + ef[e * 3 + 1] * 8 + ef[e * 3 + 2];
}

// unique combo embedding: ec[c,d] = sum of 3 bond embeddings   (512 x 16)
__global__ void combo_emb(const float* __restrict__ bemb, float* __restrict__ ec)
{
    int idx = blockIdx.x * blockDim.x + threadIdx.x;
    if (idx >= 512 * 16) return;
    int c = idx >> 4, d = idx & 15;
    int c0 = c >> 6, c1 = (c >> 3) & 7, c2 = c & 7;
    ec[idx] = bemb[(0 * 8 + c0) * 16 + d]
            + bemb[(1 * 8 + c1) * 16 + d]
            + bemb[(2 * 8 + c2) * 16 + d];
}

// edge-MLP layer 1 (K=16 too small for WMMA): h1 = relu(ec @ gW1), f16 out
__global__ void mlp1(const float* __restrict__ ec, const float* __restrict__ gW1,
                     _Float16* __restrict__ h1)
{
    int idx = blockIdx.x * blockDim.x + threadIdx.x;
    if (idx >= 512 * 1024) return;
    int c = idx >> 10, j = idx & 1023;
    float s = 0.0f;
#pragma unroll
    for (int k = 0; k < 16; ++k) s += ec[c * 16 + k] * gW1[k * 1024 + j];
    h1[idx] = (_Float16)fmaxf(s, 0.0f);
}

// NNConv message + scatter-add. One wave per edge; lane = output column.
// out[] was pre-initialized with x@root + conv_bias by the root GEMM.
__global__ void edge_msg(const float* __restrict__ xf, const float* __restrict__ w,
                         const int* __restrict__ srcv, const int* __restrict__ dstv,
                         const int* __restrict__ combo, float* __restrict__ out)
{
    int t = blockIdx.x * blockDim.x + threadIdx.x;
    int e = t >> 5, o = t & 31;
    if (e >= EE) return;
    const float* xr = xf + (size_t)srcv[e] * DIN;
    const float* wm = w + (size_t)combo[e] * (DIN * DOUT) + o;
    float acc = 0.0f;
#pragma unroll
    for (int k = 0; k < DIN; ++k) acc += xr[k] * wm[k * DOUT];
    atomicAdd(out + (size_t)dstv[e] * DOUT + o, acc);
}

// to_dense_batch -> f16 [G, 1632] (zero-padded rows 40..50)
__global__ void build_dense(const float* __restrict__ out, _Float16* __restrict__ dense)
{
    int idx = blockIdx.x * blockDim.x + threadIdx.x;
    if (idx >= GG * FDIM) return;
    int g = idx / FDIM, r = idx % FDIM;
    int n = r >> 5, o = r & 31;
    float v = (n < NPG) ? out[((size_t)g * NPG + n) * DOUT + o] : 0.0f;
    dense[idx] = (_Float16)v;
}

// readout tail: 128 -> 32 -> 8 -> 1, one wave per graph (8 graphs / block)
__global__ void tail_mlp(const _Float16* __restrict__ f2,
                         const float* __restrict__ mW3, const float* __restrict__ mb3,
                         const float* __restrict__ mW4, const float* __restrict__ mb4,
                         const float* __restrict__ mW5, const float* __restrict__ mb5,
                         float* __restrict__ out)
{
    __shared__ float buf[8][40];
    int lane = threadIdx.x & 31;
    int wib  = threadIdx.x >> 5;
    int g    = blockIdx.x * 8 + wib;

    const _Float16* f = f2 + (size_t)g * 128;
    float s = mb3[lane];
#pragma unroll 4
    for (int k = 0; k < 128; ++k) s += (float)f[k] * mW3[k * 32 + lane];
    buf[wib][lane] = fmaxf(s, 0.0f);
    __syncthreads();

    if (lane < 8) {
        float t = mb4[lane];
#pragma unroll
        for (int k = 0; k < 32; ++k) t += buf[wib][k] * mW4[k * 8 + lane];
        buf[wib][32 + lane] = fmaxf(t, 0.0f);
    }
    __syncthreads();

    if (lane == 0) {
        float t = mb5[0];
#pragma unroll
        for (int k = 0; k < 8; ++k) t += buf[wib][32 + k] * mW5[k];
        out[g] = t;
    }
}

// ---------------------------------------------------------------------------
static inline int gemm_grid(int M, int N) {
    int waves = (M >> 4) * (N >> 5);        // 16x32 tile per wave
    return (waves * 32 + 127) / 128;        // 128 threads = 4 waves per block
}

extern "C" void kernel_launch(void* const* d_in, const int* in_sizes, int n_in,
                              void* d_out, int out_size, void* d_ws, size_t ws_size,
                              hipStream_t stream)
{
    const int*   node_features = (const int*)d_in[0];
    const int*   edge_features = (const int*)d_in[1];
    const int*   edge_index    = (const int*)d_in[2];
    const float* atom_emb  = (const float*)d_in[4];
    const float* bond_emb  = (const float*)d_in[5];
    const float* gW1       = (const float*)d_in[6];
    const float* gW2       = (const float*)d_in[7];
    const float* gW3       = (const float*)d_in[8];
    const float* root      = (const float*)d_in[9];
    const float* conv_bias = (const float*)d_in[10];
    const float* mW1 = (const float*)d_in[11];
    const float* mb1 = (const float*)d_in[12];
    const float* mW2 = (const float*)d_in[13];
    const float* mb2 = (const float*)d_in[14];
    const float* mW3 = (const float*)d_in[15];
    const float* mb3 = (const float*)d_in[16];
    const float* mW4 = (const float*)d_in[17];
    const float* mb4 = (const float*)d_in[18];
    const float* mW5 = (const float*)d_in[19];
    const float* mb5 = (const float*)d_in[20];

    const int* srcv = edge_index;          // edge_index[0,:]
    const int* dstv = edge_index + EE;     // edge_index[1,:]

    // --- workspace carve-up (all sizes multiples of 256B) ---
    char* p = (char*)d_ws;
    float*    xf    = (float*)p;      p += (size_t)NN * DIN * 4;        // node feats f32
    _Float16* xh    = (_Float16*)p;   p += (size_t)NN * DIN * 2;        // node feats f16
    int*      combo = (int*)p;        p += (size_t)EE * 4;              // edge combo ids
    float*    ec    = (float*)p;      p += 512 * 16 * 4;                // combo embeddings
    _Float16* h1h   = (_Float16*)p;   p += 512 * 1024 * 2;              // edge MLP h1
    _Float16* h2h   = (_Float16*)p;   p += 512 * 256 * 2;               // edge MLP h2
    float*    wf    = (float*)p;      p += 512 * 2048 * 4;              // per-combo W (64x32)
    float*    outv  = (float*)p;      p += (size_t)NN * DOUT * 4;       // conv output
    _Float16* dense = (_Float16*)p;   p += (size_t)GG * FDIM * 2;       // dense batch
    _Float16* f1    = (_Float16*)p;   p += (size_t)GG * 256 * 2;        // readout h1
    _Float16* f2    = (_Float16*)p;   p += (size_t)GG * 128 * 2;        // readout h2
    _Float16* gW2p  = (_Float16*)p;   p += 1024 * 256 * 2;              // packed B frags
    _Float16* gW3p  = (_Float16*)p;   p += 256 * 2048 * 2;
    _Float16* rootp = (_Float16*)p;   p += DIN * DOUT * 2;
    _Float16* mW1p  = (_Float16*)p;   p += FDIM * 256 * 2;
    _Float16* mW2p  = (_Float16*)p;   p += 256 * 128 * 2;

    // --- weight conversion + WMMA-fragment packing (f32 -> f16, one pass) ---
    pack_b_f16<<<(1024 * 256 + 255) / 256, 256, 0, stream>>>(gW2, gW2p, 1024, 256);
    pack_b_f16<<<(256 * 2048 + 255) / 256, 256, 0, stream>>>(gW3, gW3p, 256, 2048);
    pack_b_f16<<<(DIN * DOUT + 255) / 256, 256, 0, stream>>>(root, rootp, DIN, DOUT);
    pack_b_f16<<<(FDIM * 256 + 255) / 256, 256, 0, stream>>>(mW1, mW1p, FDIM, 256);
    pack_b_f16<<<(256 * 128 + 255) / 256, 256, 0, stream>>>(mW2, mW2p, 256, 128);

    // --- encoders ---
    atom_enc<<<(NN * DIN + 255) / 256, 256, 0, stream>>>(node_features, atom_emb, xf, xh);
    edge_combo<<<(EE + 255) / 256, 256, 0, stream>>>(edge_features, combo);
    combo_emb<<<(512 * 16 + 255) / 256, 256, 0, stream>>>(bond_emb, ec);

    // --- edge-conditioned weight MLP on 512 unique combos ---
    mlp1<<<(512 * 1024 + 255) / 256, 256, 0, stream>>>(ec, gW1, h1h);
    gemm_wmma_f16<<<gemm_grid(512, 256), 128, 0, stream>>>(      // h2 = relu(h1 @ gW2)
        h1h, gW2p, nullptr, nullptr, h2h, 512, 256, 1024, 1);
    gemm_wmma_f16<<<gemm_grid(512, 2048), 128, 0, stream>>>(     // W  = h2 @ gW3
        h2h, gW3p, nullptr, wf, nullptr, 512, 2048, 256, 0);

    // --- NNConv: out = x @ root + bias, then scatter-add messages ---
    gemm_wmma_f16<<<gemm_grid(NN, DOUT), 128, 0, stream>>>(
        xh, rootp, conv_bias, outv, nullptr, NN, DOUT, DIN, 0);
    edge_msg<<<(EE * 32 + 255) / 256, 256, 0, stream>>>(xf, wf, srcv, dstv, combo, outv);

    // --- readout ---
    build_dense<<<(GG * FDIM + 255) / 256, 256, 0, stream>>>(outv, dense);
    gemm_wmma_f16<<<gemm_grid(GG, 256), 128, 0, stream>>>(       // relu(f @ mW1 + mb1)
        dense, mW1p, mb1, nullptr, f1, GG, 256, FDIM, 1);
    gemm_wmma_f16<<<gemm_grid(GG, 128), 128, 0, stream>>>(       // relu(f1 @ mW2 + mb2)
        f1, mW2p, mb2, nullptr, f2, GG, 128, 256, 1);
    tail_mlp<<<GG / 8, 256, 0, stream>>>(f2, mW3, mb3, mW4, mb4, mW5, mb5, (float*)d_out);
}
